// Int4WeightInt8ActLinear_18107582120173
// MI455X (gfx1250) — 8-bit matrix operand order repaired and checked by reading, compiled, not run on hardware
//
#include <hip/hip_runtime.h>

#define Mdim 4096
#define Ndim 4096
#define Kdim 4096
#define BM 128
#define BN 128
#define BK 64
#define XSTR 80
#define WSTR 80
#define NKIT (Kdim / BK)
#define XST3 (3 * BM * XSTR)

typedef int v8i __attribute__((ext_vector_type(8)));
typedef float v4f __attribute__((ext_vector_type(4)));
typedef int v4i_t __attribute__((vector_size(16)));

#if defined(__has_builtin)
#if __has_builtin(__builtin_amdgcn_global_load_async_to_lds_b128)
#define HAVE_ASYNC_B128 1
#else
#define HAVE_ASYNC_B128 0
#endif
#else
#define HAVE_ASYNC_B128 0
#endif

#if HAVE_ASYNC_B128
typedef __attribute__((address_space(1))) v4i_t GV4;
typedef __attribute__((address_space(3))) v4i_t LV4;
#endif

template <int N>
static __device__ __forceinline__ void wait_async() {
#if HAVE_ASYNC_B128
#if __has_builtin(__builtin_amdgcn_s_wait_asynccnt)
  __builtin_amdgcn_s_wait_asynccnt(N);
#else
  if (N == 0) asm volatile("s_wait_asynccnt 0" ::: "memory");
  else        asm volatile("s_wait_asynccnt 2" ::: "memory");
#endif
#endif
}

static __device__ __forceinline__ void wait_ds0() {
#if __has_builtin(__builtin_amdgcn_s_wait_dscnt)
  __builtin_amdgcn_s_wait_dscnt(0);
#else
  asm volatile("s_wait_dscnt 0" ::: "memory");
#endif
}

__global__ __launch_bounds__(256)
void quant_rowsum_kernel(const float* __restrict__ x,
                         const float* __restrict__ a_scale_p,
                         signed char* __restrict__ xq,
                         int* __restrict__ rowsum) {
  const int row = blockIdx.x;
  const int tid = threadIdx.x;
  const float a = a_scale_p[0];

  const v4f* xv = (const v4f*)(x + (size_t)row * Kdim + tid * 16);
  union { uint4 u4; signed char c[16]; } o;
  int s = 0;
#pragma unroll
  for (int j = 0; j < 4; ++j) {
    v4f f = __builtin_nontemporal_load(xv + j);
#pragma unroll
    for (int k = 0; k < 4; ++k) {
      float q = rintf(f[k] / a);
      q = fminf(127.0f, fmaxf(-127.0f, q));
      int qi = (int)q;
      o.c[j * 4 + k] = (signed char)qi;
      s += qi;
    }
  }
  *(uint4*)(xq + (size_t)row * Kdim + tid * 16) = o.u4;

  __shared__ int red[256];
  red[tid] = s;
  __syncthreads();
#pragma unroll
  for (int off = 128; off > 0; off >>= 1) {
    if (tid < off) red[tid] += red[tid + off];
    __syncthreads();
  }
  if (tid == 0) rowsum[row] = red[0];
}

__global__ __launch_bounds__(256)
void i4i8_gemm_wmma_kernel(const signed char* __restrict__ xq,
                           const int* __restrict__ rowsum,
                           const unsigned char* __restrict__ wpk,
                           const float* __restrict__ bias,
                           const float* __restrict__ w_scale_p,
                           const float* __restrict__ a_scale_p,
                           float* __restrict__ out) {
  __shared__ __align__(16) unsigned char smem[XST3 + 2 * BN * WSTR];

  const int tid  = threadIdx.x;
  const int lane = tid & 31;
  const int wave = tid >> 5;
  const int wm = wave & 1;
  const int wn = wave >> 1;
  const int mBase = blockIdx.x * BM;
  const int nBase = blockIdx.y * BN;

  const int xrow  = tid >> 1;
  const int xcol  = (tid & 1) * 32;
  const int wrow  = tid >> 1;
  const int wpcol = (tid & 1) * 16;

  const signed char*   xg = xq  + (size_t)(mBase + xrow) * Kdim + xcol;
  const unsigned char* wg = wpk + (size_t)(nBase + wrow) * (Kdim / 2) + wpcol;

  const unsigned xbase = (unsigned)(xrow * XSTR + xcol);
  const unsigned wbase = (unsigned)XST3 + (unsigned)(wrow * WSTR + wpcol * 2);

  auto issue_x = [&](int k0, int st) {
    const unsigned dst = (unsigned)(st * BM * XSTR) + xbase;
#if HAVE_ASYNC_B128
    __builtin_amdgcn_global_load_async_to_lds_b128((GV4*)(xg + k0),
                                                   (LV4*)(smem + dst), 0, 0);
    __builtin_amdgcn_global_load_async_to_lds_b128((GV4*)(xg + k0 + 16),
                                                   (LV4*)(smem + dst + 16), 0, 0);
#else
    *(uint4*)(smem + dst)      = *(const uint4*)(xg + k0);
    *(uint4*)(smem + dst + 16) = *(const uint4*)(xg + k0 + 16);
#endif
  };

  auto store_w = [&](uint4 p, int st) {
    const unsigned base = wbase + (unsigned)(st * BN * WSTR);
    unsigned pk[4] = {p.x, p.y, p.z, p.w};
    unsigned d[8];
#pragma unroll
    for (int j = 0; j < 4; ++j) {
      unsigned e = ((pk[j])      & 0x0F0F0F0Fu) ^ 0x08080808u;
      unsigned o = ((pk[j] >> 4) & 0x0F0F0F0Fu) ^ 0x08080808u;
      d[2 * j + 0] = __builtin_amdgcn_perm(o, e, 0x05010400u);
      d[2 * j + 1] = __builtin_amdgcn_perm(o, e, 0x07030602u);
    }
    *(uint4*)(smem + base)      = make_uint4(d[0], d[1], d[2], d[3]);
    *(uint4*)(smem + base + 16) = make_uint4(d[4], d[5], d[6], d[7]);
  };

  auto loadA = [&](int st, int mi) -> v8i {
    const int r = wm * 64 + mi * 16 + (lane & 15);
    const unsigned kof = (unsigned)((lane >> 4) << 4);
    const unsigned char* p = smem + (unsigned)(st * BM * XSTR) + (unsigned)(r * XSTR) + kof;
    union { v8i v; unsigned long long q[4]; } u;
    u.q[0] = *(const unsigned long long*)(p +  0);
    u.q[1] = *(const unsigned long long*)(p +  8);
    u.q[2] = *(const unsigned long long*)(p + 32);
    u.q[3] = *(const unsigned long long*)(p + 40);
    return u.v;
  };

  auto loadB = [&](int st, int ni) -> v8i {
    const int r = wn * 32 + ni * 16 + (lane & 15);
    const unsigned kof = (unsigned)((lane >> 4) << 4);
    const unsigned char* p = smem + (unsigned)XST3 + (unsigned)(st * BN * WSTR)
                           + (unsigned)(r * WSTR) + kof;
    union { v8i v; uint4 q[2]; } u;
    u.q[0] = *(const uint4*)(p);
    u.q[1] = *(const uint4*)(p + 32);
    return u.v;
  };

  v8i acc[4][2] = {};

  issue_x(0, 0);
  issue_x(BK, 1);
  uint4 wB = *(const uint4*)(wg);
  store_w(wB, 0);
  wB = *(const uint4*)(wg + (BK >> 1));
  wait_async<2>();
  wait_ds0();
  __syncthreads();

  int stC = 0;
  for (int kt = 0; kt < NKIT; ++kt) {
    const int wsC = kt & 1;
    const int wsN = wsC ^ 1;
    const bool more1 = (kt + 1) < NKIT;
    const bool more2 = (kt + 2) < NKIT;

    uint4 wNext;
    if (more2) {
      int stI = stC + 2; if (stI >= 3) stI -= 3;
      issue_x((kt + 2) * BK, stI);
      wNext = *(const uint4*)(wg + (((kt + 2) * BK) >> 1));
    }

    v8i a0 = loadA(stC, 0), a1 = loadA(stC, 1), a2 = loadA(stC, 2), a3 = loadA(stC, 3);
    v8i b0 = loadB(wsC, 0), b1 = loadB(wsC, 1);

    acc[0][0] = __builtin_amdgcn_wmma_i32_16x16x64_iu8(true, a0, false, b0, acc[0][0], false, false);
    acc[1][0] = __builtin_amdgcn_wmma_i32_16x16x64_iu8(true, a1, false, b0, acc[1][0], false, false);
    acc[2][0] = __builtin_amdgcn_wmma_i32_16x16x64_iu8(true, a2, false, b0, acc[2][0], false, false);
    acc[3][0] = __builtin_amdgcn_wmma_i32_16x16x64_iu8(true, a3, false, b0, acc[3][0], false, false);
    acc[0][1] = __builtin_amdgcn_wmma_i32_16x16x64_iu8(true, a0, false, b1, acc[0][1], false, false);
    acc[1][1] = __builtin_amdgcn_wmma_i32_16x16x64_iu8(true, a1, false, b1, acc[1][1], false, false);
    acc[2][1] = __builtin_amdgcn_wmma_i32_16x16x64_iu8(true, a2, false, b1, acc[2][1], false, false);
    acc[3][1] = __builtin_amdgcn_wmma_i32_16x16x64_iu8(true, a3, false, b1, acc[3][1], false, false);

    if (more1) store_w(wB, wsN);
    if (more2) wB = wNext;

    if (more2) wait_async<2>();
    else       wait_async<0>();
    wait_ds0();
    __syncthreads();

    stC = (stC + 1 == 3) ? 0 : stC + 1;
  }

  const float sc = a_scale_p[0] * w_scale_p[0];
  const int rowHalf = (lane >> 4) << 3;
  const int nLane = lane & 15;
#pragma unroll
  for (int mi = 0; mi < 4; ++mi) {
    const int r0 = mBase + wm * 64 + mi * 16 + rowHalf;
    int rs[8];
#pragma unroll
    for (int v = 0; v < 8; ++v) rs[v] = rowsum[r0 + v];
#pragma unroll
    for (int ni = 0; ni < 2; ++ni) {
      const int c = nBase + wn * 32 + ni * 16 + nLane;
      const float bb = bias[c];
#pragma unroll
      for (int v = 0; v < 8; ++v) {
        const int t = acc[mi][ni][v] - 8 * rs[v];
        float y = (float)t * sc + bb;
        y = (float)(_Float16)y;
        __builtin_nontemporal_store(y, &out[(size_t)(r0 + v) * Ndim + c]);
      }
    }
  }
}

extern "C" void kernel_launch(void* const* d_in, const int* in_sizes, int n_in,
                              void* d_out, int out_size, void* d_ws, size_t ws_size,
                              hipStream_t stream) {
  (void)in_sizes; (void)n_in; (void)out_size; (void)ws_size;
  const float*         x   = (const float*)d_in[0];
  const unsigned char* wp  = (const unsigned char*)d_in[1];
  const float*         bia = (const float*)d_in[2];
  const float*         wsc = (const float*)d_in[3];
  const float*         asc = (const float*)d_in[4];

  signed char* xq = (signed char*)d_ws;
  int* rowsum     = (int*)((char*)d_ws + (size_t)Mdim * Kdim);

  quant_rowsum_kernel<<<Mdim, 256, 0, stream>>>(x, asc, xq, rowsum);

  dim3 grid(Mdim / BM, Ndim / BN);
  i4i8_gemm_wmma_kernel<<<grid, 256, 0, stream>>>(xq, rowsum, wp, bia, wsc, asc,
                                                  (float*)d_out);
}
